// MultiTimeSeriesForecaster_14276471292388
// MI455X (gfx1250) — matrix operand order repaired and checked by reading, not compiled, not run on hardware
//
#include <hip/hip_runtime.h>
#include <hip/hip_bf16.h>

// ---------------------------------------------------------------------------
// MultiTimeSeriesForecaster forward for gfx1250 (MI455X), wave32 + WMMA bf16.
// B=128, S=112, HD=512, FFN=768, NH=8, HDIM=64, L=8, NP=256, MAXLEN=2000
// ~500 GFLOP of GEMMs; weights (~25MB) + activations (~29MB) are L2-resident
// (192MB L2) -> compute-bound; all contractions via v_wmma_f32_16x16x32_bf16.
// GEMM: 128x64 block tile, 32x32 wave tile (4 WMMAs / wave / k-step),
// double-buffered LDS (1 barrier per k-step), all fragments = 2x ds_load_b128
// from swizzle-padded LDS; f32->bf16 RNE conversion on the staging path.
// ---------------------------------------------------------------------------

#define B_   128
#define S_   112
#define HD_  512
#define FFN_ 768
#define NH_  8
#define HDIM_ 64
#define L_   8
#define NP_  256
#define MAXLEN_ 2000
#define M_   (B_ * S_)            // 14336 = 112 * 128
#define EPS_ 1e-5f

typedef __attribute__((ext_vector_type(16))) __bf16 v16bf;
typedef __attribute__((ext_vector_type(8)))  __bf16 v8bf;
typedef __attribute__((ext_vector_type(8)))  float  v8f;
typedef __attribute__((ext_vector_type(4)))  unsigned short u16x4;
typedef __attribute__((ext_vector_type(8)))  unsigned short u16x8;

// float -> bf16 bits with round-to-nearest-even
__device__ __forceinline__ unsigned short f2bfu(float f) {
  union { float f; unsigned u; } x; x.f = f;
  unsigned u = x.u;
  unsigned r = (u + 0x7FFFu + ((u >> 16) & 1u)) >> 16;
  return (unsigned short)r;
}
__device__ __forceinline__ v8f wmma_bf16(v16bf a, v16bf b, v8f c) {
  return __builtin_amdgcn_wmma_f32_16x16x32_bf16(false, a, false, b, (short)0, c,
                                                 false, false);
}
// Build a 16-elem bf16 fragment from two contiguous 16-byte LDS runs.
__device__ __forceinline__ v16bf frag16(const unsigned short* p0,
                                        const unsigned short* p1) {
  v8bf lo = *(const v8bf*)p0;
  v8bf hi = *(const v8bf*)p1;
  return __builtin_shufflevector(lo, hi, 0, 1, 2, 3, 4, 5, 6, 7,
                                         8, 9, 10, 11, 12, 13, 14, 15);
}
__device__ __forceinline__ float gelu_exact(float x) {
  return 0.5f * x * (1.0f + erff(x * 0.70710678118654752f));
}

// ---------------------------------------------------------------------------
// Generic GEMM: C[M,N] = act(A[M,K] @ W + bias)
//   transW==0 : W stored [K,N] ; transW==1 : W stored [N,K]
//   act: 0=none, 1=gelu(exact)
// 256 threads (8 waves). Block tile 128x64, BK=32, double-buffered LDS.
// Wave w: rows (w>>1)*32, cols (w&1)*32 -> 2 A-frags x 2 B-frags -> 4 WMMAs.
// aS[m][k], bS[n][k] K-contiguous, rows padded to 40 ushorts (20-DWORD pitch
// -> 16 distinct banks across lanes 0..15, 16B-aligned fragment runs).
// ---------------------------------------------------------------------------
#define GBM 128
#define GBN 64
#define GBK 32
#define GPITCH 40

__global__ __launch_bounds__(256)
void gemm_bias_kernel(const float* __restrict__ A, const float* __restrict__ W,
                      const float* __restrict__ bias, float* __restrict__ C,
                      int M, int N, int K, int transW, int act)
{
  __shared__ unsigned short aS[2][GBM][GPITCH];
  __shared__ unsigned short bS[2][GBN][GPITCH];

  const int tid  = threadIdx.x;
  const int bm   = (int)blockIdx.y * GBM;
  const int bn   = (int)blockIdx.x * GBN;
  const int wave = tid >> 5;
  const int lane = tid & 31;

  const int wm   = (wave >> 1) * 32;       // 0,32,64,96
  const int wn   = (wave & 1) * 32;        // 0,32
  const int lrow = lane & 15;
  const int akb  = (lane & 16) ? 16 : 0;   // A k-sub-base (elements)
  const int bkb  = (lane & 16) ? 16 : 0;   // B k-sub-base (elements)
  const int off8 = (lane & 16) ? 8 : 0;

  v8f acc00 = {}, acc01 = {}, acc10 = {}, acc11 = {};

  // ---- staging helpers (each stages one 32-wide K slice into buffer `buf`) --
  auto stageA = [&](int buf, int kk0) {
    const int r = tid >> 1;                // 0..127
    const int c = (tid & 1) * 16;          // 0 or 16
    const float* ga = A + (size_t)(bm + r) * K + (kk0 + c);
    if (kk0 + GBK < K) __builtin_prefetch(ga + GBK, 0, 1);
    u16x8 v0, v1;
    #pragma unroll
    for (int j = 0; j < 8; ++j) { v0[j] = f2bfu(ga[j]); v1[j] = f2bfu(ga[8 + j]); }
    *(u16x8*)&aS[buf][r][c]     = v0;
    *(u16x8*)&aS[buf][r][c + 8] = v1;
  };
  auto stageB = [&](int buf, int kk0) {
    if (!transW) {
      // W[k][n]: each thread covers 4 k-rows x 2 n-cols -> two 8B DS stores
      const int n2 = (tid & 31) * 2;       // 0..62
      const int k4 = (tid >> 5) * 4;       // 0..28
      u16x4 c0, c1;
      #pragma unroll
      for (int kk = 0; kk < 4; ++kk) {
        const float* gw = W + (size_t)(kk0 + k4 + kk) * N + (bn + n2);
        c0[kk] = f2bfu(gw[0]);
        c1[kk] = f2bfu(gw[1]);
      }
      if (kk0 + GBK < K)
        __builtin_prefetch(W + (size_t)(kk0 + GBK + k4) * N + (bn + n2), 0, 1);
      *(u16x4*)&bS[buf][n2][k4]     = c0;
      *(u16x4*)&bS[buf][n2 + 1][k4] = c1;
    } else {
      // W[n][k]: contiguous k run -> 16B DS store
      const int n = tid >> 2;              // 0..63
      const int c = (tid & 3) * 8;         // k offset
      const float* gw = W + (size_t)(bn + n) * K + (kk0 + c);
      if (kk0 + GBK < K) __builtin_prefetch(gw + GBK, 0, 1);
      u16x8 bv;
      #pragma unroll
      for (int j = 0; j < 8; ++j) bv[j] = f2bfu(gw[j]);
      *(u16x8*)&bS[buf][n][c] = bv;
    }
  };

  stageA(0, 0);
  stageB(0, 0);
  __syncthreads();

  int cur = 0;
  for (int k0 = 0; k0 < K; k0 += GBK) {
    if (k0 + GBK < K) { stageA(cur ^ 1, k0 + GBK); stageB(cur ^ 1, k0 + GBK); }

    const v16bf af0 = frag16(&aS[cur][wm + lrow][akb],
                             &aS[cur][wm + lrow][akb + 8]);
    const v16bf af1 = frag16(&aS[cur][wm + 16 + lrow][akb],
                             &aS[cur][wm + 16 + lrow][akb + 8]);
    const v16bf bf0 = frag16(&bS[cur][wn + lrow][bkb],
                             &bS[cur][wn + lrow][bkb + 8]);
    const v16bf bf1 = frag16(&bS[cur][wn + 16 + lrow][bkb],
                             &bS[cur][wn + 16 + lrow][bkb + 8]);
    acc00 = wmma_bf16(af0, bf0, acc00);
    acc01 = wmma_bf16(af0, bf1, acc01);
    acc10 = wmma_bf16(af1, bf0, acc10);
    acc11 = wmma_bf16(af1, bf1, acc11);

    __syncthreads();          // publishes next buffer, protects current one
    cur ^= 1;
  }

  // ---- epilogue: bias + activation + store (4 16x16 tiles per wave) ----
  const int ccol0 = bn + wn + lrow;
  const float b0 = bias ? bias[ccol0]      : 0.0f;
  const float b1 = bias ? bias[ccol0 + 16] : 0.0f;
  #pragma unroll
  for (int r = 0; r < 8; ++r) {
    const size_t row0 = (size_t)(bm + wm + off8 + r);
    const size_t row1 = row0 + 16;
    float v00 = acc00[r] + b0, v01 = acc01[r] + b1;
    float v10 = acc10[r] + b0, v11 = acc11[r] + b1;
    if (act == 1) {
      v00 = gelu_exact(v00); v01 = gelu_exact(v01);
      v10 = gelu_exact(v10); v11 = gelu_exact(v11);
    }
    C[row0 * N + ccol0]      = v00;
    C[row0 * N + ccol0 + 16] = v01;
    C[row1 * N + ccol0]      = v10;
    C[row1 * N + ccol0 + 16] = v11;
  }
}

// ---------------------------------------------------------------------------
// Attention: one block per (batch, head, 16-query tile). QK^T and PV via WMMA.
// kS/qS: [s][d] padded to 72 (36-DWORD pitch). vT: V transposed [d][s] padded
// to 136 (68-DWORD pitch), s zero-padded 112->128. pS: bf16-packed softmax
// probabilities [16][136], zero-padded. All fragments = two ds_load_b128.
// ---------------------------------------------------------------------------
#define KPITCH 72
#define VPITCH 136

__global__ __launch_bounds__(256)
void attn_kernel(const float* __restrict__ Qg, const float* __restrict__ Kg,
                 const float* __restrict__ Vg, const float* __restrict__ rpb,
                 const unsigned char* __restrict__ mask, float* __restrict__ Og)
{
  const int qt   = blockIdx.x;   // 0..6
  const int h    = blockIdx.y;   // 0..7
  const int bIdx = blockIdx.z;   // 0..127
  const int tid  = threadIdx.x;
  const int wave = tid >> 5;
  const int lane = tid & 31;

  __shared__ unsigned short kS[S_][KPITCH];
  __shared__ unsigned short qS[16][KPITCH];
  __shared__ unsigned short vT[HDIM_][VPITCH];
  __shared__ unsigned short pS[16][VPITCH];
  __shared__ float          sS[16][S_];

  const size_t baseRow = (size_t)bIdx * S_;
  const size_t headOff = (size_t)h * HDIM_;

  // ---- stage K (112x64) and Q (16x64), 8-wide chunks, 16B DS stores ----
  for (int i = tid; i < S_ * 8; i += 256) {        // 896 chunks
    const int s = i >> 3, d = (i & 7) * 8;
    const float* gp = Kg + (baseRow + s) * HD_ + headOff + d;
    u16x8 vv;
    #pragma unroll
    for (int j = 0; j < 8; ++j) vv[j] = f2bfu(gp[j]);
    *(u16x8*)&kS[s][d] = vv;
  }
  for (int i = tid; i < 16 * 8; i += 256) {        // 128 chunks
    const int s = i >> 3, d = (i & 7) * 8;
    const float* gp = Qg + (baseRow + qt * 16 + s) * HD_ + headOff + d;
    u16x8 vv;
    #pragma unroll
    for (int j = 0; j < 8; ++j) vv[j] = f2bfu(gp[j]);
    *(u16x8*)&qS[s][d] = vv;
  }
  // ---- stage V transposed: vT[d][s], zero-padded s>=112 ----
  for (int i = tid; i < HDIM_ * 16; i += 256) {    // 1024 chunks
    const int d = i >> 4, s0 = (i & 15) * 8;
    u16x8 vv;
    #pragma unroll
    for (int j = 0; j < 8; ++j) {
      const int s = s0 + j;
      vv[j] = (s < S_) ? f2bfu(Vg[(baseRow + s) * HD_ + headOff + d]) : 0;
    }
    *(u16x8*)&vT[d][s0] = vv;
  }
  __syncthreads();

  const int lrow = lane & 15;
  const int akb  = (lane & 16) ? 16 : 0;
  const int bkb  = (lane & 16) ? 16 : 0;
  const int off8 = (lane & 16) ? 8 : 0;

  // ---- scores = Q.K^T * scale + rpb, masked ----
  if (wave < 7) {
    const int jt = wave;
    v8f acc = {};
    #pragma unroll
    for (int kk = 0; kk < HDIM_; kk += 32) {
      const v16bf af  = frag16(&qS[lrow][kk + akb], &qS[lrow][kk + akb + 8]);
      const v16bf bfr = frag16(&kS[jt * 16 + lrow][kk + bkb],
                               &kS[jt * 16 + lrow][kk + bkb + 8]);
      acc = wmma_bf16(af, bfr, acc);
    }
    const int jj = jt * 16 + lrow;
    const bool mOK = mask[baseRow + jj] != 0;
    #pragma unroll
    for (int r = 0; r < 8; ++r) {
      const int i  = r + off8;
      const int qi = qt * 16 + i;
      float val = acc[r] * 0.125f
                + rpb[(size_t)(qi - jj + (MAXLEN_ - 1)) * NH_ + h];
      sS[i][jj] = mOK ? val : -1e30f;
    }
  }
  __syncthreads();

  // ---- row softmax (16 rows) ----
  if (tid < 16) {
    float mx = -3.4e38f;
    for (int j = 0; j < S_; ++j) mx = fmaxf(mx, sS[tid][j]);
    float sum = 0.0f;
    for (int j = 0; j < S_; ++j) { float e = expf(sS[tid][j] - mx); sS[tid][j] = e; sum += e; }
    const float inv = 1.0f / sum;
    for (int j = 0; j < S_; ++j) sS[tid][j] *= inv;
  }
  __syncthreads();

  // ---- pack P to bf16, zero-padded to K=128 ----
  {
    const int r  = tid >> 4;            // 0..15
    const int c0 = (tid & 15) * 8;      // 0..120
    u16x8 vv;
    #pragma unroll
    for (int j = 0; j < 8; ++j) {
      const int c = c0 + j;
      vv[j] = (c < S_) ? f2bfu(sS[r][c]) : 0;
    }
    *(u16x8*)&pS[r][c0] = vv;
  }
  __syncthreads();

  // ---- out = P.V ----
  if (wave < 4) {
    const int nt = wave;
    v8f acc = {};
    #pragma unroll
    for (int kk = 0; kk < 128; kk += 32) {
      const v16bf af  = frag16(&pS[lrow][kk + akb], &pS[lrow][kk + akb + 8]);
      const v16bf bfr = frag16(&vT[nt * 16 + lrow][kk + bkb],
                               &vT[nt * 16 + lrow][kk + bkb + 8]);
      acc = wmma_bf16(af, bfr, acc);
    }
    #pragma unroll
    for (int r = 0; r < 8; ++r) {
      const int i = r + off8;
      const size_t go = (baseRow + qt * 16 + i) * HD_
                      + headOff + nt * 16 + lrow;
      Og[go] = acc[r];
    }
  }
}

// ---------------------------------------------------------------------------
// Fused residual + LayerNorm over 512 channels. One block (256 thr) per row.
// Y = LN(X + R) * g + b   (R may be null). Safe in-place (Y==X).
// ---------------------------------------------------------------------------
__global__ __launch_bounds__(256)
void ln512_kernel(const float* __restrict__ X, const float* __restrict__ R,
                  const float* __restrict__ g, const float* __restrict__ b,
                  float* __restrict__ Y)
{
  const int row = blockIdx.x, t = threadIdx.x;
  const float* x = X + (size_t)row * HD_;
  float v0 = x[t], v1 = x[t + 256];
  if (R) { const float* r = R + (size_t)row * HD_; v0 += r[t]; v1 += r[t + 256]; }

  __shared__ float red[256];
  red[t] = v0 + v1; __syncthreads();
  for (int o = 128; o > 0; o >>= 1) { if (t < o) red[t] += red[t + o]; __syncthreads(); }
  const float mean = red[0] * (1.0f / HD_);
  __syncthreads();
  const float d0 = v0 - mean, d1 = v1 - mean;
  red[t] = d0 * d0 + d1 * d1; __syncthreads();
  for (int o = 128; o > 0; o >>= 1) { if (t < o) red[t] += red[t + o]; __syncthreads(); }
  const float inv = rsqrtf(red[0] * (1.0f / HD_) + EPS_);
  Y[(size_t)row * HD_ + t]       = d0 * inv * g[t]       + b[t];
  Y[(size_t)row * HD_ + t + 256] = d1 * inv * g[t + 256] + b[t + 256];
}

// ---------------------------------------------------------------------------
// Masked softmax over NP=256 prototypes. One block (256 thr) per row, in-place.
// ---------------------------------------------------------------------------
__global__ __launch_bounds__(256)
void softmax256_kernel(float* __restrict__ P, const unsigned char* __restrict__ mask)
{
  const int row = blockIdx.x, t = threadIdx.x;
  float v = P[(size_t)row * NP_ + t] * (mask[row] ? 1.0f : 0.0f);
  __shared__ float red[256];
  red[t] = v; __syncthreads();
  for (int o = 128; o > 0; o >>= 1) { if (t < o) red[t] = fmaxf(red[t], red[t + o]); __syncthreads(); }
  const float mx = red[0]; __syncthreads();
  const float e = expf(v - mx);
  red[t] = e; __syncthreads();
  for (int o = 128; o > 0; o >>= 1) { if (t < o) red[t] += red[t + o]; __syncthreads(); }
  P[(size_t)row * NP_ + t] = e / red[0];
}

// ---------------------------------------------------------------------------
// Depthwise conv k=5 along S (channels-last layout [B,S,HD]).
// ---------------------------------------------------------------------------
__global__ __launch_bounds__(256)
void dwconv_kernel(const float* __restrict__ X, const float* __restrict__ w,
                   const float* __restrict__ bias, float* __restrict__ Y)
{
  const long long gidx = (long long)blockIdx.x * 256 + threadIdx.x;
  const long long total = (long long)M_ * HD_;
  if (gidx >= total) return;
  const int c = (int)(gidx & (HD_ - 1));
  const long long bs = gidx >> 9;
  const int s = (int)(bs % S_);
  float acc = bias[c];
  #pragma unroll
  for (int t = 0; t < 5; ++t) {
    const int ss = s + t - 2;
    if (ss >= 0 && ss < S_)
      acc += w[c * 5 + t] * X[gidx + (long long)(t - 2) * HD_];
  }
  Y[gidx] = acc;
}

// ---------------------------------------------------------------------------
// Embedding branch: conv1d(2->128, k) + LayerNorm(128) + SiLU -> Hcat columns.
// ---------------------------------------------------------------------------
__global__ __launch_bounds__(128)
void embed_branch_kernel(const float* __restrict__ X, const float* __restrict__ w,
                         const float* __restrict__ bias, const float* __restrict__ g,
                         const float* __restrict__ be, float* __restrict__ Hcat,
                         int ksize, int coff)
{
  const int bs = blockIdx.x;
  const int sPos = bs % S_, bIdx = bs / S_;
  const int q = threadIdx.x;
  const int pad = (ksize - 1) >> 1;

  float acc = bias[q];
  for (int t = 0; t < ksize; ++t) {
    const int ss = sPos + t - pad;
    if (ss >= 0 && ss < S_) {
      const float* xp = X + ((size_t)bIdx * S_ + ss) * 2;
      acc += w[(q * 2 + 0) * ksize + t] * xp[0]
           + w[(q * 2 + 1) * ksize + t] * xp[1];
    }
  }
  __shared__ float red[128];
  red[q] = acc; __syncthreads();
  for (int o = 64; o > 0; o >>= 1) { if (q < o) red[q] += red[q + o]; __syncthreads(); }
  const float mean = red[0] * (1.0f / 128.0f);
  __syncthreads();
  const float d = acc - mean;
  red[q] = d * d; __syncthreads();
  for (int o = 64; o > 0; o >>= 1) { if (q < o) red[q] += red[q + o]; __syncthreads(); }
  const float inv = rsqrtf(red[0] * (1.0f / 128.0f) + EPS_);
  const float hn = d * inv * g[q] + be[q];
  const float sv = hn / (1.0f + expf(-hn));   // silu
  Hcat[(size_t)bs * HD_ + coff + q] = sv;
}

// ---------------------------------------------------------------------------
// Host-side orchestration
// ---------------------------------------------------------------------------
static inline void launch_gemm(const float* A, const float* W, const float* bias,
                               float* C, int M, int N, int K, int transW, int act,
                               hipStream_t s)
{
  dim3 grid(N / GBN, M / GBM);
  gemm_bias_kernel<<<grid, 256, 0, s>>>(A, W, bias, C, M, N, K, transW, act);
}

extern "C" void kernel_launch(void* const* d_in, const int* in_sizes, int n_in,
                              void* d_out, int out_size, void* d_ws, size_t ws_size,
                              hipStream_t stream)
{
  (void)in_sizes; (void)n_in; (void)out_size; (void)ws_size;

  // ---- input index map (setup_inputs() dict insertion order) ----
  const float*         X    = (const float*)d_in[0];          // [B,S,2]
  const unsigned char* mask = (const unsigned char*)d_in[1];  // [B,S] bool
  const float* ew[4]  = {(const float*)d_in[2],  (const float*)d_in[6],
                         (const float*)d_in[10], (const float*)d_in[14]};
  const float* eb[4]  = {(const float*)d_in[3],  (const float*)d_in[7],
                         (const float*)d_in[11], (const float*)d_in[15]};
  const float* eg[4]  = {(const float*)d_in[4],  (const float*)d_in[8],
                         (const float*)d_in[12], (const float*)d_in[16]};
  const float* ebe[4] = {(const float*)d_in[5],  (const float*)d_in[9],
                         (const float*)d_in[13], (const float*)d_in[17]};
  const int    eks[4] = {3, 7, 15, 31};
  const float* proj_w = (const float*)d_in[18];
  const float* proj_b = (const float*)d_in[19];
  const float* fn_g   = (const float*)d_in[20];
  const float* fn_b   = (const float*)d_in[21];
  const float* dw_w = (const float*)d_in[22];  const float* dw_b = (const float*)d_in[23];
  const float* pw_w = (const float*)d_in[24];  const float* pw_b = (const float*)d_in[25];
  const float* wq = (const float*)d_in[26];    const float* bq = (const float*)d_in[27];
  const float* wk = (const float*)d_in[28];    const float* bk = (const float*)d_in[29];
  const float* wv = (const float*)d_in[30];    const float* bv = (const float*)d_in[31];
  const float* wo = (const float*)d_in[32];    const float* bo = (const float*)d_in[33];
  const float* rpb = (const float*)d_in[34];
  const float* n1g = (const float*)d_in[35];   const float* n1b = (const float*)d_in[36];
  const float* n2g = (const float*)d_in[37];   const float* n2b = (const float*)d_in[38];
  const float* n3g = (const float*)d_in[39];   const float* n3b = (const float*)d_in[40];
  const float* f1w = (const float*)d_in[41];   const float* f1b = (const float*)d_in[42];
  const float* f2w = (const float*)d_in[43];   const float* f2b = (const float*)d_in[44];
  const float* mn1g = (const float*)d_in[45];  const float* mn1b = (const float*)d_in[46];
  const float* m1w  = (const float*)d_in[47];  const float* m1b  = (const float*)d_in[48];
  const float* m2w  = (const float*)d_in[49];  const float* m2b  = (const float*)d_in[50];
  const float* bank = (const float*)d_in[51];
  const float* mpw  = (const float*)d_in[52];  const float* mpb  = (const float*)d_in[53];
  const float* mn2g = (const float*)d_in[54];  const float* mn2b = (const float*)d_in[55];

  float* out = (float*)d_out;

  // ---- workspace carve (all fp32) ----
  float* ws   = (float*)d_ws;
  float* hbuf = ws;                              // [M,512]
  float* t1   = hbuf + (size_t)M_ * HD_;         // [M,768] (also Hcat)
  float* t2   = t1   + (size_t)M_ * FFN_;        // [M,768]
  float* qb   = t2   + (size_t)M_ * FFN_;        // [M,512] (also pw logits)
  float* kb   = qb   + (size_t)M_ * HD_;
  float* vb   = kb   + (size_t)M_ * HD_;
  float* att  = vb   + (size_t)M_ * HD_;

  // ================= embedding =================
  for (int br = 0; br < 4; ++br)
    embed_branch_kernel<<<M_, 128, 0, stream>>>(X, ew[br], eb[br], eg[br], ebe[br],
                                                t1, eks[br], br * 128);
  launch_gemm(t1, proj_w, proj_b, t2, M_, HD_, HD_, 0, 0, stream);
  ln512_kernel<<<M_, 256, 0, stream>>>(t2, nullptr, fn_g, fn_b, hbuf);

  // ================= transformer blocks =================
  for (int l = 0; l < L_; ++l) {
    const float* dww = dw_w + (size_t)l * HD_ * 5;
    const float* dwb = dw_b + (size_t)l * HD_;
    const float* pww = pw_w + (size_t)l * HD_ * HD_;
    const float* pwb = pw_b + (size_t)l * HD_;
    const float* lwq = wq + (size_t)l * HD_ * HD_;  const float* lbq = bq + (size_t)l * HD_;
    const float* lwk = wk + (size_t)l * HD_ * HD_;  const float* lbk = bk + (size_t)l * HD_;
    const float* lwv = wv + (size_t)l * HD_ * HD_;  const float* lbv = bv + (size_t)l * HD_;
    const float* lwo = wo + (size_t)l * HD_ * HD_;  const float* lbo = bo + (size_t)l * HD_;
    const float* lrpb = rpb + (size_t)l * (2 * MAXLEN_ - 1) * NH_;
    const float* l1g = n1g + (size_t)l * HD_;  const float* l1b = n1b + (size_t)l * HD_;
    const float* l2g = n2g + (size_t)l * HD_;  const float* l2b = n2b + (size_t)l * HD_;
    const float* l3g = n3g + (size_t)l * HD_;  const float* l3b = n3b + (size_t)l * HD_;
    const float* lf1w = f1w + (size_t)l * HD_ * FFN_; const float* lf1b = f1b + (size_t)l * FFN_;
    const float* lf2w = f2w + (size_t)l * FFN_ * HD_; const float* lf2b = f2b + (size_t)l * HD_;

    dwconv_kernel<<<(M_ * HD_ + 255) / 256, 256, 0, stream>>>(hbuf, dww, dwb, t1);
    launch_gemm(t1, pww, pwb, t2, M_, HD_, HD_, 1, 0, stream);   // pw_w is [O,I]
    ln512_kernel<<<M_, 256, 0, stream>>>(hbuf, t2, l1g, l1b, hbuf);

    launch_gemm(hbuf, lwq, lbq, qb, M_, HD_, HD_, 0, 0, stream);
    launch_gemm(hbuf, lwk, lbk, kb, M_, HD_, HD_, 0, 0, stream);
    launch_gemm(hbuf, lwv, lbv, vb, M_, HD_, HD_, 0, 0, stream);
    attn_kernel<<<dim3(S_ / 16, NH_, B_), 256, 0, stream>>>(qb, kb, vb, lrpb, mask, att);
    launch_gemm(att, lwo, lbo, t1, M_, HD_, HD_, 0, 0, stream);
    ln512_kernel<<<M_, 256, 0, stream>>>(hbuf, t1, l2g, l2b, hbuf);

    launch_gemm(hbuf, lf1w, lf1b, t1, M_, FFN_, HD_, 0, 1, stream);   // gelu
    launch_gemm(t1, lf2w, lf2b, t2, M_, HD_, FFN_, 0, 0, stream);
    ln512_kernel<<<M_, 256, 0, stream>>>(hbuf, t2, l3g, l3b, hbuf);
  }

  // ================= memory module =================
  ln512_kernel<<<M_, 256, 0, stream>>>(hbuf, nullptr, mn1g, mn1b, t1);
  launch_gemm(t1, m1w, m1b, t2, M_, HD_, HD_, 0, 1, stream);          // gelu
  launch_gemm(t2, m2w, m2b, qb, M_, NP_, HD_, 0, 0, stream);          // [M,256]
  softmax256_kernel<<<M_, 256, 0, stream>>>(qb, mask);
  launch_gemm(qb, bank, nullptr, t1, M_, HD_, NP_, 0, 0, stream);     // pw @ bank
  launch_gemm(t1, mpw, mpb, t2, M_, HD_, HD_, 0, 0, stream);          // @ m['pw'] + pb
  ln512_kernel<<<M_, 256, 0, stream>>>(hbuf, t2, mn2g, mn2b, out);
}
